// GroupedQueryAttentionWithSink_44813688767042
// MI455X (gfx1250) — compile-verified, matrix operand order checked by reading, not run on hardware
//
#include <hip/hip_runtime.h>
#include <hip/hip_bf16.h>

typedef __attribute__((ext_vector_type(16))) _Float16 v16h;
typedef __attribute__((ext_vector_type(8)))  float    v8f;
typedef __attribute__((ext_vector_type(4)))  int      v4i;

#define S_LEN 2048
#define D_IN 2048
#define NH 32
#define NKV 8
#define HD 64

#if defined(__gfx1250__) && __has_builtin(__builtin_amdgcn_global_load_async_to_lds_b128)
#define HAVE_ASYNC 1
#else
#define HAVE_ASYNC 0
#endif

union F8 { v8f v; float f[8]; };

__device__ __forceinline__ v8f wmma32(v16h a, v16h b, v8f c) {
  return __builtin_amdgcn_wmma_f32_16x16x32_f16(false, a, false, b, (short)0, c, false, false);
}

// Load a 16-half A/B fragment per ISA layout: elements 0..7 = p[0..7], 8..15 = p[16..23].
__device__ __forceinline__ v16h ld_frag(const _Float16* p) {
  union { v16h v; _Float16 e[16]; } u;
#pragma unroll
  for (int i = 0; i < 8; ++i) { u.e[i] = p[i]; u.e[8 + i] = p[16 + i]; }
  return u.v;
}

// 16-byte global -> LDS copy; async (ASYNCcnt) when the builtin exists.
__device__ __forceinline__ void cp_async16(const _Float16* g, _Float16* l) {
#if HAVE_ASYNC
  __builtin_amdgcn_global_load_async_to_lds_b128(
      (__attribute__((address_space(1))) v4i*)g,
      (__attribute__((address_space(3))) v4i*)l, 0, 0);
#else
  *(uint4*)l = *(const uint4*)g;
#endif
}

__device__ __forceinline__ void wait_async_le2() {
#if HAVE_ASYNC
#if __has_builtin(__builtin_amdgcn_s_wait_asynccnt)
  __builtin_amdgcn_s_wait_asynccnt(2);
#else
  asm volatile("s_wait_asynccnt 0x2" ::: "memory");
#endif
#endif
}

__device__ __forceinline__ void wait_async_le0() {
#if HAVE_ASYNC
#if __has_builtin(__builtin_amdgcn_s_wait_asynccnt)
  __builtin_amdgcn_s_wait_asynccnt(0);
#else
  asm volatile("s_wait_asynccnt 0x0" ::: "memory");
#endif
#endif
}

// ---------------- f32 -> f16 convert ----------------
__global__ void cvt_f16_kernel(const float* __restrict__ a, _Float16* __restrict__ b, int n) {
  int i = blockIdx.x * blockDim.x + threadIdx.x;
  if (i < n) b[i] = (_Float16)a[i];
}

// ---------------- WMMA GEMM: C[M,N] = A[M,K](f16) * B[K,N](f16), f32 out ----------------
// Double-buffered LDS; A staged via async-to-LDS, B transposed through registers so the
// global loads issue before the WMMA block (latency hidden) and scatter-stores after it.
__global__ __launch_bounds__(256) void gemm_f16_kernel(const _Float16* __restrict__ A,
                                                       const _Float16* __restrict__ B,
                                                       float* __restrict__ C,
                                                       int M, int N, int K) {
  __shared__ _Float16 As[2][128][48];   // padded stride: 96B rows -> conflict-spread b128 reads
  __shared__ _Float16 Bs[2][128][48];   // transposed: Bs[buf][n][k]

  const int t = threadIdx.x;
  const int lane = t & 31;
  const int wave = t >> 5;
  const int m0 = blockIdx.y * 128;
  const int n0 = blockIdx.x * 128;
  const int wm = (wave >> 1) * 32;   // wave covers 32 rows
  const int wn = (wave & 1) * 64;    // and 64 cols

  F8 acc[2][4];
#pragma unroll
  for (int i = 0; i < 2; ++i)
#pragma unroll
    for (int j = 0; j < 4; ++j) acc[i][j].v = (v8f){};

  const int arow = t >> 1, acol = (t & 1) * 16;
  const int bk = t >> 3,  bn = (t & 7) * 16;
  const int lo = (lane & 16) ? 8 : 0;
  const int rsel = lane & 15;

  auto issueA = [&](int kt, int b) {
    const _Float16* ga = A + (size_t)(m0 + arow) * K + kt * 32 + acol;
    cp_async16(ga,     &As[b][arow][acol]);
    cp_async16(ga + 8, &As[b][arow][acol + 8]);
  };
  auto loadB = [&](int kt, uint4& r0, uint4& r1) {
    const _Float16* gb = B + (size_t)(kt * 32 + bk) * N + n0 + bn;
    r0 = *(const uint4*)gb;
    r1 = *(const uint4*)(gb + 8);
  };
  auto storeB = [&](int b, uint4 r0, uint4 r1) {
    union { uint4 u; _Float16 h[8]; } x0, x1;
    x0.u = r0; x1.u = r1;
#pragma unroll
    for (int i = 0; i < 8; ++i) {
      Bs[b][bn + i][bk]     = x0.h[i];
      Bs[b][bn + 8 + i][bk] = x1.h[i];
    }
  };

  const int T = K >> 5;
  // prologue: tile 0 into buffer 0
  issueA(0, 0);
  {
    uint4 r0, r1;
    loadB(0, r0, r1);
    storeB(0, r0, r1);
  }

  int buf = 0;
  for (int kt = 0; kt < T; ++kt) {
    const bool hn = (kt + 1) < T;
    uint4 nr0, nr1;
    if (hn) {
      issueA(kt + 1, buf ^ 1);          // async into the other buffer
      loadB(kt + 1, nr0, nr1);          // global loads in flight across the WMMA block
      wait_async_le2();                 // current tile's A asyncs complete
    } else {
      wait_async_le0();
    }
    __syncthreads();

    v16h af[2], bf[4];
#pragma unroll
    for (int im = 0; im < 2; ++im) af[im] = ld_frag(&As[buf][wm + im * 16 + rsel][lo]);
#pragma unroll
    for (int in = 0; in < 4; ++in) bf[in] = ld_frag(&Bs[buf][wn + in * 16 + rsel][lo]);
#pragma unroll
    for (int im = 0; im < 2; ++im)
#pragma unroll
      for (int in = 0; in < 4; ++in)
        acc[im][in].v = wmma32(af[im], bf[in], acc[im][in].v);
    __syncthreads();

    if (hn) storeB(buf ^ 1, nr0, nr1);
    buf ^= 1;
  }

  const int rg = (lane >> 4) * 8;
#pragma unroll
  for (int im = 0; im < 2; ++im)
#pragma unroll
    for (int in = 0; in < 4; ++in) {
      float* cp = C + (size_t)(m0 + wm + im * 16 + rg) * N + n0 + wn + in * 16 + rsel;
#pragma unroll
      for (int e = 0; e < 8; ++e) cp[(size_t)e * N] = acc[im][in].f[e];
    }
}

// ---------------- fused RMSNorm + RoPE, reformat to [head][s][d] f16 ----------------
__global__ __launch_bounds__(256) void norm_rope_kernel(const float* __restrict__ proj, // [S, nh*64]
                                                        const float* __restrict__ cosT,
                                                        const float* __restrict__ sinT,
                                                        const float* __restrict__ w,
                                                        _Float16* __restrict__ outHead,
                                                        int nh) {
  const int wave = threadIdx.x >> 5;
  const int lane = threadIdx.x & 31;
  const int rowId = blockIdx.x * 8 + wave;       // s*nh + h
  const int s = rowId / nh, h = rowId % nh;
  const float* p = proj + (size_t)s * (nh * HD) + h * HD;
  float x1 = p[lane], x2 = p[lane + 32];
  float ss = x1 * x1 + x2 * x2;
#pragma unroll
  for (int off = 16; off; off >>= 1) ss += __shfl_xor(ss, off, 32);
  float r = rsqrtf(ss * (1.0f / 64.0f) + 1e-6f);
  float n1 = x1 * r * w[lane], n2 = x2 * r * w[lane + 32];
  float c1 = cosT[s * HD + lane], s1 = sinT[s * HD + lane];
  float c2 = cosT[s * HD + lane + 32], s2 = sinT[s * HD + lane + 32];
  _Float16* o = outHead + ((size_t)h * S_LEN + s) * HD;
  o[lane]      = (_Float16)(n1 * c1 - n2 * s1);
  o[lane + 32] = (_Float16)(n2 * c2 + n1 * s2);
}

// ---------------- V transpose: [s][kvh*64+d] f32 -> [kvh*64+d][s] f16 ----------------
__global__ void vtrans_kernel(const float* __restrict__ vf, _Float16* __restrict__ vt) {
  int i = blockIdx.x * blockDim.x + threadIdx.x;   // S*NKV*64
  int s = i >> 9, c = i & 511;
  vt[(size_t)c * S_LEN + s] = (_Float16)vf[i];
}

// ---------------- flash attention w/ sink: one wave per (head, 16-row q tile) ----------------
__global__ __launch_bounds__(32) void attn_kernel(const _Float16* __restrict__ qh, // [h][s][d]
                                                  const _Float16* __restrict__ kh, // [kvh][s][d]
                                                  const _Float16* __restrict__ vt, // [kvh*64+d][s]
                                                  const float* __restrict__ sink,
                                                  _Float16* __restrict__ ao) {    // [s][h*64+d]
  __shared__ _Float16 P[16][32];
  const int lane = threadIdx.x;
  const int h = blockIdx.y;
  const int qb = blockIdx.x * 16;
  const int kvh = h >> 2;
  const int rsel = lane & 15;
  const int lo = (lane & 16) ? 8 : 0;
  const int rg = (lane >> 4) * 8;

  const _Float16* qp = qh + ((size_t)h * S_LEN + qb + rsel) * HD;
  v16h qf0 = ld_frag(qp + lo);        // d 0..31
  v16h qf1 = ld_frag(qp + 32 + lo);   // d 32..63

  F8 o[4];
#pragma unroll
  for (int d = 0; d < 4; ++d) o[d].v = (v8f){};
  float m[8], l[8];
#pragma unroll
  for (int e = 0; e < 8; ++e) { m[e] = -3.0e38f; l[e] = 0.0f; }

  const int ktiles = (qb + 16 + 31) >> 5;
  for (int kt = 0; kt < ktiles; ++kt) {
    const int kb = kt * 32;
    const _Float16* kp0 = kh + ((size_t)kvh * S_LEN + kb + rsel) * HD + lo;
    const _Float16* kp1 = kh + ((size_t)kvh * S_LEN + kb + 16 + rsel) * HD + lo;
    if (kt + 1 < ktiles) {
      __builtin_prefetch(kp0 + 32 * HD, 0, 0);
      __builtin_prefetch(kp1 + 32 * HD, 0, 0);
    }
    F8 s0, s1; s0.v = (v8f){}; s1.v = (v8f){};
    s0.v = wmma32(qf0, ld_frag(kp0),      s0.v);
    s0.v = wmma32(qf1, ld_frag(kp0 + 32), s0.v);
    s1.v = wmma32(qf0, ld_frag(kp1),      s1.v);
    s1.v = wmma32(qf1, ld_frag(kp1 + 32), s1.v);

    float mnew[8];
#pragma unroll
    for (int e = 0; e < 8; ++e) {
      const int row = qb + rg + e;
      const int c0 = kb + rsel, c1 = c0 + 16;
      float a = s0.f[e] * 0.125f; if (c0 > row) a = -1.0e30f;
      float b = s1.f[e] * 0.125f; if (c1 > row) b = -1.0e30f;
      s0.f[e] = a; s1.f[e] = b;
      float mx = fmaxf(a, b);
      mx = fmaxf(mx, __shfl_xor(mx, 8, 16));
      mx = fmaxf(mx, __shfl_xor(mx, 4, 16));
      mx = fmaxf(mx, __shfl_xor(mx, 2, 16));
      mx = fmaxf(mx, __shfl_xor(mx, 1, 16));
      mnew[e] = fmaxf(m[e], mx);
    }
#pragma unroll
    for (int e = 0; e < 8; ++e) {
      const float sc = __expf(m[e] - mnew[e]);
      const float p0 = __expf(s0.f[e] - mnew[e]);
      const float p1 = __expf(s1.f[e] - mnew[e]);
      float rs = p0 + p1;
      rs += __shfl_xor(rs, 8, 16);
      rs += __shfl_xor(rs, 4, 16);
      rs += __shfl_xor(rs, 2, 16);
      rs += __shfl_xor(rs, 1, 16);
      l[e] = l[e] * sc + rs;
      m[e] = mnew[e];
#pragma unroll
      for (int d = 0; d < 4; ++d) o[d].f[e] *= sc;
      P[rg + e][rsel]      = (_Float16)p0;
      P[rg + e][rsel + 16] = (_Float16)p1;
    }
    __syncthreads();
    v16h pf = ld_frag(&P[rsel][lo]);
#pragma unroll
    for (int d = 0; d < 4; ++d) {
      const _Float16* vp = vt + ((size_t)(kvh * HD + d * 16 + rsel)) * S_LEN + kb + lo;
      o[d].v = wmma32(pf, ld_frag(vp), o[d].v);
    }
    __syncthreads();
  }

  const float sk = sink[h];
  float denom[8];
#pragma unroll
  for (int e = 0; e < 8; ++e) denom[e] = l[e] + __expf(sk - m[e]);
#pragma unroll
  for (int d = 0; d < 4; ++d)
#pragma unroll
    for (int e = 0; e < 8; ++e)
      ao[(size_t)(qb + rg + e) * (NH * HD) + h * HD + d * 16 + rsel] =
          (_Float16)(o[d].f[e] / denom[e]);
}

extern "C" void kernel_launch(void* const* d_in, const int* in_sizes, int n_in,
                              void* d_out, int out_size, void* d_ws, size_t ws_size,
                              hipStream_t stream) {
  (void)in_sizes; (void)n_in; (void)out_size; (void)ws_size;
  const float* x    = (const float*)d_in[0];
  // d_in[1] = mask (computed analytically, unused)
  const float* cosT = (const float*)d_in[2];
  const float* sinT = (const float*)d_in[3];
  const float* wq   = (const float*)d_in[4];
  const float* wk   = (const float*)d_in[5];
  const float* wv   = (const float*)d_in[6];
  const float* wo   = (const float*)d_in[7];
  const float* qnw  = (const float*)d_in[8];
  const float* knw  = (const float*)d_in[9];
  const float* sink = (const float*)d_in[10];
  float* out = (float*)d_out;

  char* ws = (char*)d_ws;
  const size_t MB = 1ull << 20;
  _Float16* xh   = (_Float16*)(ws + 0 * MB);    // 8 MiB
  _Float16* wqh  = (_Float16*)(ws + 8 * MB);    // 8 MiB
  _Float16* wkh  = (_Float16*)(ws + 16 * MB);   // 2 MiB
  _Float16* wvh  = (_Float16*)(ws + 18 * MB);   // 2 MiB
  _Float16* woh  = (_Float16*)(ws + 20 * MB);   // 8 MiB
  float*    qf   = (float*)   (ws + 28 * MB);   // 16 MiB
  float*    kf   = (float*)   (ws + 44 * MB);   // 4 MiB
  float*    vf   = (float*)   (ws + 48 * MB);   // 4 MiB
  _Float16* qhh  = (_Float16*)(ws + 52 * MB);   // 8 MiB
  _Float16* khh  = (_Float16*)(ws + 60 * MB);   // 2 MiB
  _Float16* vth  = (_Float16*)(ws + 62 * MB);   // 2 MiB
  _Float16* ao   = (_Float16*)(ws + 64 * MB);   // 8 MiB

  // 1) f32 -> f16 casts
  cvt_f16_kernel<<<(S_LEN * D_IN + 255) / 256, 256, 0, stream>>>(x, xh, S_LEN * D_IN);
  cvt_f16_kernel<<<(D_IN * NH * HD + 255) / 256, 256, 0, stream>>>(wq, wqh, D_IN * NH * HD);
  cvt_f16_kernel<<<(D_IN * NKV * HD + 255) / 256, 256, 0, stream>>>(wk, wkh, D_IN * NKV * HD);
  cvt_f16_kernel<<<(D_IN * NKV * HD + 255) / 256, 256, 0, stream>>>(wv, wvh, D_IN * NKV * HD);
  cvt_f16_kernel<<<(NH * HD * D_IN + 255) / 256, 256, 0, stream>>>(wo, woh, NH * HD * D_IN);

  // 2) projections (WMMA GEMMs)
  gemm_f16_kernel<<<dim3(16, 16), 256, 0, stream>>>(xh, wqh, qf, S_LEN, NH * HD, D_IN);
  gemm_f16_kernel<<<dim3(4, 16),  256, 0, stream>>>(xh, wkh, kf, S_LEN, NKV * HD, D_IN);
  gemm_f16_kernel<<<dim3(4, 16),  256, 0, stream>>>(xh, wvh, vf, S_LEN, NKV * HD, D_IN);

  // 3) RMSNorm + RoPE + per-head layout
  norm_rope_kernel<<<(S_LEN * NH) / 8, 256, 0, stream>>>(qf, cosT, sinT, qnw, qhh, NH);
  norm_rope_kernel<<<(S_LEN * NKV) / 8, 256, 0, stream>>>(kf, cosT, sinT, knw, khh, NKV);
  vtrans_kernel<<<(S_LEN * NKV * HD) / 256, 256, 0, stream>>>(vf, vth);

  // 4) flash attention with sink
  attn_kernel<<<dim3(S_LEN / 16, NH), 32, 0, stream>>>(qhh, khh, vth, sink, ao);

  // 5) output projection (WMMA GEMM) -> f32 d_out
  gemm_f16_kernel<<<dim3(16, 16), 256, 0, stream>>>(ao, woh, out, S_LEN, D_IN, NH * HD);
}
